// NASPolicy_4552665334253
// MI455X (gfx1250) — matrix operand order repaired and checked by reading, not compiled, not run on hardware
//
#include <hip/hip_runtime.h>

// ---------------------------------------------------------------------------
// NAS controller LSTM (20 steps, CELL=128, batch=1) as a single persistent
// workgroup on one WGP.
//   Phase 0: build X[32][128] (f16) in LDS from embeddings (rows 20..31 = 0)
//   Phase 1: U = X @ W_ih^T + b_ih + b_hh   (one-shot WMMA GEMM, U -> LDS)
//   Phase 2: 20x { gates = U[s] + h @ W_hh^T (WMMA, W_hh held in VGPRs);
//                  LSTM nonlinearity; 6-logit head; softmax -> out }
// ---------------------------------------------------------------------------

typedef _Float16 v16h __attribute__((ext_vector_type(16)));
typedef float    v8f  __attribute__((ext_vector_type(8)));

#define CELL   128
#define STEPS  20
#define NOUT   6
#define G4     (4 * CELL)      // 512 gate outputs
#define KTILES (CELL / 32)     // 4 k-tiles of 32
#define THREADS 512            // 16 wave32 waves

__global__ __launch_bounds__(THREADS)
void nas_policy_kernel(const int*   __restrict__ net,
                       const float* __restrict__ emb_start,
                       const float* __restrict__ emb_keys,   // [4,6,128]
                       const float* __restrict__ fcW,        // [4,6,128]
                       const float* __restrict__ fcb,        // [4,6]
                       const float* __restrict__ Wih,        // [512,128]
                       const float* __restrict__ Whh,        // [512,128]
                       const float* __restrict__ bih,        // [512]
                       const float* __restrict__ bhh,        // [512]
                       float*       __restrict__ out)        // [20,1,6]
{
    __shared__ _Float16 Xs[32][CELL];        // 8 KB  : padded input matrix
    __shared__ float    Us[STEPS][G4];       // 40 KB : precomputed input-gates
    __shared__ float    Gs[G4];              // 2 KB  : current-step gates
    __shared__ _Float16 h16[CELL];           // h for WMMA A operand
    __shared__ float    h32[CELL];           // h for the fp32 head
    __shared__ float    Ls[NOUT];            // logits

    const int tid    = threadIdx.x;
    const int lane   = tid & 31;
    const int wave   = tid >> 5;             // 0..15, owns N-tiles 2w, 2w+1
    const int grp    = lane >> 4;            // half-wave (0 or 1)
    const int nlocal = lane & 15;

    // Prefetch W_hh (256 KB) into cache while phase 0/1 run.
    #pragma unroll
    for (int i = 0; i < 4; ++i)
        __builtin_prefetch(Whh + (size_t)(tid * 4 + i) * 32, 0, 1);

    // ---- Phase 0: X (f16) in LDS ------------------------------------------
    for (int idx = tid; idx < 32 * CELL; idx += THREADS) {
        int row = idx >> 7, col = idx & (CELL - 1);
        float v = 0.f;
        if (row == 0) {
            v = emb_start[col];
        } else if (row < STEPS) {
            int t   = row - 1;
            int key = t & 3;                 // key_idx[t] = t % 4
            int a   = net[t];
            v = emb_keys[(key * NOUT + a) * CELL + col];
        }
        Xs[row][col] = (_Float16)v;
    }
    if (tid < CELL) { h16[tid] = (_Float16)0.f; h32[tid] = 0.f; }
    __syncthreads();

    // ---- Phase 1: U = X @ Wih^T + (bih + bhh) -----------------------------
    // B-operand tile, contiguous k order used here (A matches): element e of v16h in lane l holds
    // B[k][n] with n = nlocal, k = grp*16 + e  (VGPR r packs rows 2r, 2r+1).
    v16h Bt[2][KTILES];
    #pragma unroll
    for (int nti = 0; nti < 2; ++nti) {
        int n = (2 * wave + nti) * 16 + nlocal;
        const float* wrow = Wih + (size_t)n * CELL;
        #pragma unroll
        for (int kt = 0; kt < KTILES; ++kt) {
            const float* p = wrow + kt * 32 + grp * 16;
            v16h b;
            #pragma unroll
            for (int e = 0; e < 16; ++e) b[e] = (_Float16)p[e];
            Bt[nti][kt] = b;
        }
    }
    #pragma unroll
    for (int nti = 0; nti < 2; ++nti) {
        int nt = 2 * wave + nti;
        int n  = nt * 16 + nlocal;
        float bias = bih[n] + bhh[n];
        #pragma unroll
        for (int mt = 0; mt < 2; ++mt) {
            v8f acc;
            #pragma unroll
            for (int r = 0; r < 8; ++r) acc[r] = bias;   // bias[n] on every row
            #pragma unroll
            for (int kt = 0; kt < KTILES; ++kt) {
                // A-operand, contiguous k order as in B: lanes 0-15 hold K 0-15,
                // lanes 16-31 hold K 16-31; m = nlocal.
                v16h a;
                const _Float16* xr = &Xs[mt * 16 + nlocal][kt * 32];
                #pragma unroll
                for (int e = 0; e < 16; ++e) {
                    int k = grp * 16 + e;
                    a[e] = xr[k];
                }
                acc = __builtin_amdgcn_wmma_f32_16x16x32_f16(
                          false, a, false, Bt[nti][kt], (short)0, acc, false, false);
            }
            // D layout: VGPR r, lanes 0-15 -> M=r, lanes 16-31 -> M=8+r
            #pragma unroll
            for (int r = 0; r < 8; ++r) {
                int m = mt * 16 + grp * 8 + r;
                if (m < STEPS) Us[m][n] = acc[r];
            }
        }
    }

    // Overwrite register tiles with W_hh for the recurrence.
    #pragma unroll
    for (int nti = 0; nti < 2; ++nti) {
        int n = (2 * wave + nti) * 16 + nlocal;
        const float* wrow = Whh + (size_t)n * CELL;
        #pragma unroll
        for (int kt = 0; kt < KTILES; ++kt) {
            const float* p = wrow + kt * 32 + grp * 16;
            v16h b;
            #pragma unroll
            for (int e = 0; e < 16; ++e) b[e] = (_Float16)p[e];
            Bt[nti][kt] = b;
        }
    }
    float cstate = 0.f;                       // thread tid<128 owns c[tid]
    __syncthreads();

    // ---- Phase 2: sequential recurrence -----------------------------------
    for (int s = 0; s < STEPS; ++s) {
        // gates = U[s] + h @ Whh^T  (only D row 0 is meaningful; rows 1..15
        // repeat row 0 (every A row is h) and are discarded)
        #pragma unroll
        for (int nti = 0; nti < 2; ++nti) {
            int nt = 2 * wave + nti;
            int n  = nt * 16 + nlocal;
            v8f acc;
            float u = Us[s][n];
            #pragma unroll
            for (int r = 0; r < 8; ++r) acc[r] = u;
            #pragma unroll
            for (int kt = 0; kt < KTILES; ++kt) {
                v16h a;
                const _Float16* hr = &h16[kt * 32];
                #pragma unroll
                for (int e = 0; e < 16; ++e) {
                    int k = grp * 16 + e;
                    a[e] = hr[k];
                }
                acc = __builtin_amdgcn_wmma_f32_16x16x32_f16(
                          false, a, false, Bt[nti][kt], (short)0, acc, false, false);
            }
            if (lane < 16) Gs[n] = acc[0];    // row M=0 = VGPR0, lanes 0-15
        }
        __syncthreads();                      // [A] gates visible

        if (tid < CELL) {                     // LSTM nonlinearity, order i,f,g,o
            float gi = Gs[tid];
            float gf = Gs[CELL + tid];
            float gg = Gs[2 * CELL + tid];
            float go = Gs[3 * CELL + tid];
            float si = 1.f / (1.f + __expf(-gi));
            float sf = 1.f / (1.f + __expf(-gf));
            float so = 1.f / (1.f + __expf(-go));
            cstate = sf * cstate + si * tanhf(gg);
            float hv = so * tanhf(cstate);
            h32[tid] = hv;
            h16[tid] = (_Float16)hv;
        }
        __syncthreads();                      // [B] h visible

        // Head: 6 waves (tid 128..319) each reduce one logit.
        if (tid >= 128 && tid < 128 + 32 * NOUT) {
            int o   = (tid - 128) >> 5;
            int key = s & 3;
            const float* wrow = fcW + (size_t)(key * NOUT + o) * CELL;
            float part = 0.f;
            #pragma unroll
            for (int k = lane; k < CELL; k += 32) part += h32[k] * wrow[k];
            #pragma unroll
            for (int off = 16; off > 0; off >>= 1) part += __shfl_down(part, off);
            if (lane == 0) Ls[o] = part + fcb[key * NOUT + o];
        }
        __syncthreads();                      // [C] logits visible

        if (tid == 0) {                       // softmax over 6 values
            float m = Ls[0];
            #pragma unroll
            for (int i = 1; i < NOUT; ++i) m = fmaxf(m, Ls[i]);
            float e[NOUT], sum = 0.f;
            #pragma unroll
            for (int i = 0; i < NOUT; ++i) { e[i] = __expf(Ls[i] - m); sum += e[i]; }
            float inv = 1.f / sum;
            #pragma unroll
            for (int i = 0; i < NOUT; ++i) out[s * NOUT + i] = e[i] * inv;
        }
        // No extra barrier: next iteration's writers to Gs/Ls sit behind [A]/[B],
        // which happen-after every reader above.
    }
}

extern "C" void kernel_launch(void* const* d_in, const int* in_sizes, int n_in,
                              void* d_out, int out_size, void* d_ws, size_t ws_size,
                              hipStream_t stream) {
    const int*   net = (const int*)  d_in[0];
    // d_in[1] = reward (unused in forward)
    const float* es  = (const float*)d_in[2];
    const float* ek  = (const float*)d_in[3];
    const float* fw  = (const float*)d_in[4];
    const float* fb  = (const float*)d_in[5];
    const float* wih = (const float*)d_in[6];
    const float* whh = (const float*)d_in[7];
    const float* bi  = (const float*)d_in[8];
    const float* bh  = (const float*)d_in[9];
    (void)in_sizes; (void)n_in; (void)out_size; (void)d_ws; (void)ws_size;

    nas_policy_kernel<<<1, THREADS, 0, stream>>>(net, es, ek, fw, fb,
                                                 wih, whh, bi, bh,
                                                 (float*)d_out);
}
